// Decoder_67310727463442
// MI455X (gfx1250) — matrix operand order repaired and checked by reading, not compiled, not run on hardware
//
#include <hip/hip_runtime.h>
#include <hip/hip_bf16.h>
#include <math.h>

typedef __attribute__((ext_vector_type(16))) _Float16 v16h;
typedef __attribute__((ext_vector_type(8)))  _Float16 v8h;
typedef __attribute__((ext_vector_type(8)))  float    v8f;
typedef __attribute__((ext_vector_type(4)))  float    v4f;

#define DMODEL 512
#define NHEAD  8
#define DHEAD  64
#define DFF    2048
#define NLAYER 6
#define BATCH  8
#define LTGT   512
#define LSRC   512
#define VOCAB  32000
#define PADIDX 1
#define MROWS  (BATCH*LTGT)   /* 4096 */

// Per-lane fragment run padded to 24 halves (48B): keeps 16B alignment for
// ds_*_b128 and gives a 12-bank stride (<=2-way conflicts).
#define FRAG_STRIDE 24

// ---------------------------------------------------------------------------
// WMMA GEMM: C[z] = alpha * (A[z] x B[z]) + bias (+ReLU)
// Block tile 256(M) x 64(N), K-chunk 32; 8 waves, each owning TWO 16-row
// M-subtiles so every B fragment load from LDS feeds 2 WMMAs
// (12 ds_load_b128 -> 8 v_wmma per wave per K-step).
// LDS tiles are stored in WMMA *fragment-major* layout (ISA 7.12.2): each
// lane's 16 halves of an A/B fragment are contiguous.
// ---------------------------------------------------------------------------
template<bool TRANSB, bool RELU>
__global__ __launch_bounds__(256) void gemm_wmma_kernel(
    const float* __restrict__ A, const float* __restrict__ B,
    const float* __restrict__ bias, float* __restrict__ C,
    int K, int lda, int ldb, int ldc, float alpha,
    int Hb, long sAb, long sAh, long sBb, long sBh, long sCb, long sCh)
{
  __shared__ _Float16 Af[16 * 32 * FRAG_STRIDE];  // 16 M-subtiles x 32 lanes
  __shared__ _Float16 Bf[4 * 32 * FRAG_STRIDE];   // 4 N-subtiles x 32 lanes

  const int tid  = threadIdx.x;
  const int lane = tid & 31;
  const int wid  = tid >> 5;

  const int z  = blockIdx.z;
  const int zb = z / Hb;
  const int zh = z - zb * Hb;
  A += (size_t)zb * sAb + (size_t)zh * sAh;
  B += (size_t)zb * sBb + (size_t)zh * sBh;
  C += (size_t)zb * sCb + (size_t)zh * sCh;

  const int blockN = blockIdx.x * 64;
  const int blockM = blockIdx.y * 256;

  v8f acc0[4], acc1[4];
#pragma unroll
  for (int t = 0; t < 4; ++t)
#pragma unroll
    for (int r = 0; r < 8; ++r) { acc0[t][r] = 0.0f; acc1[t][r] = 0.0f; }

  const int nCol   = lane & 15;           // C/D column within subtile
  const int rowSel = (lane >> 4) * 8;     // C/D: vgpr r -> M = rowSel + r

  for (int k0 = 0; k0 < K; k0 += 32) {
    // ---- Stage A tile 256x32 into fragment layout ----
    // 1024 chunks of 8 consecutive K-values; chunk (row, koct) maps to lane
    // m + 16*(koct>>1), halves j0 = 8*(koct&1) .. j0+7  (contiguous 16B).
    for (int c = tid; c < 1024; c += 256) {
      int row = c >> 2, koct = c & 3;
      const float* src = A + (size_t)(blockM + row) * lda + (k0 + koct * 8);
      v4f f0 = *(const v4f*)src;
      v4f f1 = *(const v4f*)(src + 4);
      v8h h;
#pragma unroll
      for (int i = 0; i < 4; ++i) { h[i] = (_Float16)f0[i]; h[4 + i] = (_Float16)f1[i]; }
      int w = row >> 4, m = row & 15;
      int dlane = m + 16 * (koct >> 1);
      int j0 = 8 * (koct & 1);
      *(v8h*)&Af[(w * 32 + dlane) * FRAG_STRIDE + j0] = h;
    }
    // ---- Stage B tile 32x64 into fragment layout ----
    if (TRANSB) {
      // B[k][n] = Bm[n*ldb + k]: a global row (fixed n, 16 consecutive k)
      // is exactly one lane's 16 contiguous halves.
      if (tid < 128) {
        int n = tid >> 1, kb = tid & 1;
        const float* src = B + (size_t)(blockN + n) * ldb + (k0 + kb * 16);
        v4f f0 = *(const v4f*)src;
        v4f f1 = *(const v4f*)(src + 4);
        v4f f2 = *(const v4f*)(src + 8);
        v4f f3 = *(const v4f*)(src + 12);
        v8h h0, h1;
#pragma unroll
        for (int i = 0; i < 4; ++i) {
          h0[i] = (_Float16)f0[i]; h0[4 + i] = (_Float16)f1[i];
          h1[i] = (_Float16)f2[i]; h1[4 + i] = (_Float16)f3[i];
        }
        int base = ((n >> 4) * 32 + (n & 15) + 16 * kb) * FRAG_STRIDE;
        *(v8h*)&Bf[base]     = h0;
        *(v8h*)&Bf[base + 8] = h1;
      }
    } else {
      // Row-major B: coalesced global row reads, scattered b16 LDS stores.
      int k = tid >> 3, n8 = (tid & 7) * 8;
      const float* src = B + (size_t)(k0 + k) * ldb + (blockN + n8);
      v4f f0 = *(const v4f*)src;
      v4f f1 = *(const v4f*)(src + 4);
#pragma unroll
      for (int i = 0; i < 8; ++i) {
        int n = n8 + i;
        float f = (i < 4) ? f0[i] : f1[i - 4];
        Bf[((n >> 4) * 32 + (n & 15) + 16 * (k >> 4)) * FRAG_STRIDE + (k & 15)] =
            (_Float16)f;
      }
    }
    if (k0 + 32 < K) {  // gfx1250 global_prefetch_b8 of next A chunk
      __builtin_prefetch(A + (size_t)(blockM + (tid & 255)) * lda + (k0 + 32), 0, 0);
    }
    __syncthreads();

    // ---- Fragment loads: 2x ds_load_b128 each ----
    const int abase0 = ((wid * 2 + 0) * 32 + lane) * FRAG_STRIDE;
    const int abase1 = ((wid * 2 + 1) * 32 + lane) * FRAG_STRIDE;
    v8h a0lo = *(const v8h*)&Af[abase0];
    v8h a0hi = *(const v8h*)&Af[abase0 + 8];
    v8h a1lo = *(const v8h*)&Af[abase1];
    v8h a1hi = *(const v8h*)&Af[abase1 + 8];
    v16h afr0 = __builtin_shufflevector(a0lo, a0hi,
        0, 1, 2, 3, 4, 5, 6, 7, 8, 9, 10, 11, 12, 13, 14, 15);
    v16h afr1 = __builtin_shufflevector(a1lo, a1hi,
        0, 1, 2, 3, 4, 5, 6, 7, 8, 9, 10, 11, 12, 13, 14, 15);
#pragma unroll
    for (int nt = 0; nt < 4; ++nt) {
      const int bbase = (nt * 32 + lane) * FRAG_STRIDE;
      v8h blo = *(const v8h*)&Bf[bbase];
      v8h bhi = *(const v8h*)&Bf[bbase + 8];
      v16h bfr = __builtin_shufflevector(blo, bhi,
          0, 1, 2, 3, 4, 5, 6, 7, 8, 9, 10, 11, 12, 13, 14, 15);
      acc0[nt] = __builtin_amdgcn_wmma_f32_16x16x32_f16(
          false, afr0, false, bfr, (short)0, acc0[nt], false, false);
      acc1[nt] = __builtin_amdgcn_wmma_f32_16x16x32_f16(
          false, afr1, false, bfr, (short)0, acc1[nt], false, false);
    }
    __syncthreads();
  }

#pragma unroll
  for (int nt = 0; nt < 4; ++nt) {
    int n = blockN + nt * 16 + nCol;
    float bb = bias ? bias[n] : 0.0f;
#pragma unroll
    for (int r = 0; r < 8; ++r) {
      int m0 = blockM + (wid * 2 + 0) * 16 + rowSel + r;
      int m1 = blockM + (wid * 2 + 1) * 16 + rowSel + r;
      float v0 = acc0[nt][r] * alpha + bb;
      float v1 = acc1[nt][r] * alpha + bb;
      if (RELU) { v0 = fmaxf(v0, 0.0f); v1 = fmaxf(v1, 0.0f); }
      C[(size_t)m0 * ldc + n] = v0;
      C[(size_t)m1 * ldc + n] = v1;
    }
  }
}

// ---------------------------------------------------------------------------
// Embedding + sinusoidal positional encoding: x[b,l,d] = emb[tgt[b,l],d] + PE
// ---------------------------------------------------------------------------
__global__ __launch_bounds__(256) void embed_pe_kernel(
    const int* __restrict__ tgt, const float* __restrict__ emb,
    float* __restrict__ x)
{
  const float KPE = 9.210340371976184f / 255.0f;  // ln(1e4)/(nts-1)
  int idx = blockIdx.x * 256 + threadIdx.x;
  int d  = idx & (DMODEL - 1);
  int bl = idx >> 9;                // flat (b*LT + l)
  int l  = bl & (LTGT - 1);
  int token = tgt[bl];
  float pe;
  if (d < DMODEL / 2) {
    float inv = __expf(-(float)d * KPE);
    pe = __sinf((float)l * inv);
  } else {
    float inv = __expf(-(float)(d - DMODEL / 2) * KPE);
    pe = __cosf((float)l * inv);
  }
  x[idx] = emb[(size_t)token * DMODEL + d] + pe;
}

// ---------------------------------------------------------------------------
// Wave-per-row softmax over 512 keys with on-the-fly masks (wave32, 16/lane)
// rows = (b*H + h)*LT + q
// ---------------------------------------------------------------------------
__global__ __launch_bounds__(256) void softmax_self_kernel(
    float* __restrict__ probs, const int* __restrict__ tgt)
{
  int row  = blockIdx.x * 8 + (threadIdx.x >> 5);
  int lane = threadIdx.x & 31;
  int q  = row & (LTGT - 1);
  int bh = row >> 9;
  int b  = bh >> 3;
  size_t base = (size_t)row * LSRC;
  float v[16];
  float mx = -INFINITY;
#pragma unroll
  for (int j = 0; j < 16; ++j) {
    int k = lane + j * 32;
    float s = probs[base + k];
    bool masked = (k > q) || (tgt[b * LTGT + k] == PADIDX);
    v[j] = masked ? -INFINITY : s;
    mx = fmaxf(mx, v[j]);
  }
  for (int o = 16; o; o >>= 1) mx = fmaxf(mx, __shfl_xor(mx, o, 32));
  float sum = 0.0f;
#pragma unroll
  for (int j = 0; j < 16; ++j) { v[j] = __expf(v[j] - mx); sum += v[j]; }
  for (int o = 16; o; o >>= 1) sum += __shfl_xor(sum, o, 32);
  float inv = 1.0f / sum;
#pragma unroll
  for (int j = 0; j < 16; ++j) probs[base + lane + j * 32] = v[j] * inv;
}

__global__ __launch_bounds__(256) void softmax_cross_kernel(
    float* __restrict__ probs, const unsigned char* __restrict__ mask_src)
{
  int row  = blockIdx.x * 8 + (threadIdx.x >> 5);
  int lane = threadIdx.x & 31;
  int bh = row >> 9;
  int b  = bh >> 3;
  size_t base = (size_t)row * LSRC;
  float v[16];
  float mx = -INFINITY;
#pragma unroll
  for (int j = 0; j < 16; ++j) {
    int k = lane + j * 32;
    float s = probs[base + k];
    bool masked = mask_src[b * LSRC + k] != 0;
    v[j] = masked ? -INFINITY : s;
    mx = fmaxf(mx, v[j]);
  }
  for (int o = 16; o; o >>= 1) mx = fmaxf(mx, __shfl_xor(mx, o, 32));
  float sum = 0.0f;
#pragma unroll
  for (int j = 0; j < 16; ++j) { v[j] = __expf(v[j] - mx); sum += v[j]; }
  for (int o = 16; o; o >>= 1) sum += __shfl_xor(sum, o, 32);
  float inv = 1.0f / sum;
#pragma unroll
  for (int j = 0; j < 16; ++j) probs[base + lane + j * 32] = v[j] * inv;
}

// ---------------------------------------------------------------------------
// Coverage: cov[b,q,k] = mean over heads of cross-attn probs
// ---------------------------------------------------------------------------
__global__ __launch_bounds__(256) void cov_kernel(
    const float* __restrict__ probs, float* __restrict__ cov)
{
  int idx = blockIdx.x * 256 + threadIdx.x;
  int k = idx & (LSRC - 1);
  int q = (idx >> 9) & (LTGT - 1);
  int b = idx >> 18;
  float s = 0.0f;
#pragma unroll
  for (int h = 0; h < NHEAD; ++h)
    s += probs[((size_t)(b * NHEAD + h) * LTGT + q) * LSRC + k];
  cov[idx] = s * (1.0f / NHEAD);
}

// ---------------------------------------------------------------------------
// Fused residual add + LayerNorm, one wave per 512-wide row (16 elems/lane)
// ---------------------------------------------------------------------------
__global__ __launch_bounds__(256) void lnres_kernel(
    const float* __restrict__ a, const float* __restrict__ res,
    const float* __restrict__ g, const float* __restrict__ beta,
    float* __restrict__ out)
{
  int row  = blockIdx.x * 8 + (threadIdx.x >> 5);
  int lane = threadIdx.x & 31;
  size_t base = (size_t)row * DMODEL;
  float v[16];
  float s = 0.0f;
#pragma unroll
  for (int j = 0; j < 16; ++j) {
    int c = lane + j * 32;
    v[j] = a[base + c] + res[base + c];
    s += v[j];
  }
  for (int o = 16; o; o >>= 1) s += __shfl_xor(s, o, 32);
  float mean = s * (1.0f / DMODEL);
  float var = 0.0f;
#pragma unroll
  for (int j = 0; j < 16; ++j) { float d = v[j] - mean; var += d * d; }
  for (int o = 16; o; o >>= 1) var += __shfl_xor(var, o, 32);
  var *= (1.0f / DMODEL);
  float rstd = rsqrtf(var + 1e-5f);
#pragma unroll
  for (int j = 0; j < 16; ++j) {
    int c = lane + j * 32;
    out[base + c] = (v[j] - mean) * rstd * g[c] + beta[c];
  }
}

// ---------------------------------------------------------------------------
// In-place log-softmax over rows of 32000 (one 256-thread block per row)
// ---------------------------------------------------------------------------
__global__ __launch_bounds__(256) void logsoftmax_kernel(float* __restrict__ logits)
{
  __shared__ float red[256];
  size_t base = (size_t)blockIdx.x * VOCAB;
  float mx = -INFINITY;
  for (int c = threadIdx.x; c < VOCAB; c += 256)
    mx = fmaxf(mx, logits[base + c]);
  red[threadIdx.x] = mx;
  __syncthreads();
  for (int s2 = 128; s2; s2 >>= 1) {
    if (threadIdx.x < s2) red[threadIdx.x] = fmaxf(red[threadIdx.x], red[threadIdx.x + s2]);
    __syncthreads();
  }
  mx = red[0];
  __syncthreads();
  float sum = 0.0f;
  for (int c = threadIdx.x; c < VOCAB; c += 256)
    sum += __expf(logits[base + c] - mx);
  red[threadIdx.x] = sum;
  __syncthreads();
  for (int s2 = 128; s2; s2 >>= 1) {
    if (threadIdx.x < s2) red[threadIdx.x] += red[threadIdx.x + s2];
    __syncthreads();
  }
  float lse = __logf(red[0]) + mx;
  for (int c = threadIdx.x; c < VOCAB; c += 256)
    logits[base + c] -= lse;
}

// ---------------------------------------------------------------------------
// Host-side GEMM launcher  (block tile 256M x 64N)
// ---------------------------------------------------------------------------
static void rungemm(hipStream_t st, const float* A, const float* B, const float* bias,
                    float* C, int M, int N, int K, int lda, int ldb, int ldc,
                    float alpha, bool transB, bool relu, int nb, int Hb,
                    long sAb, long sAh, long sBb, long sBh, long sCb, long sCh)
{
  dim3 g(N / 64, M / 256, nb), blk(256, 1, 1);
  if (transB) {
    if (relu) gemm_wmma_kernel<true, true><<<g, blk, 0, st>>>(A, B, bias, C, K, lda, ldb, ldc, alpha, Hb, sAb, sAh, sBb, sBh, sCb, sCh);
    else      gemm_wmma_kernel<true, false><<<g, blk, 0, st>>>(A, B, bias, C, K, lda, ldb, ldc, alpha, Hb, sAb, sAh, sBb, sBh, sCb, sCh);
  } else {
    if (relu) gemm_wmma_kernel<false, true><<<g, blk, 0, st>>>(A, B, bias, C, K, lda, ldb, ldc, alpha, Hb, sAb, sAh, sBb, sBh, sCb, sCh);
    else      gemm_wmma_kernel<false, false><<<g, blk, 0, st>>>(A, B, bias, C, K, lda, ldb, ldc, alpha, Hb, sAb, sAh, sBb, sBh, sCb, sCh);
  }
}

extern "C" void kernel_launch(void* const* d_in, const int* in_sizes, int n_in,
                              void* d_out, int out_size, void* d_ws, size_t ws_size,
                              hipStream_t stream)
{
  (void)in_sizes; (void)n_in; (void)out_size; (void)ws_size;

  const int*           tgt      = (const int*)d_in[0];
  const float*         context  = (const float*)d_in[1];
  const unsigned char* mask_src = (const unsigned char*)d_in[2];
  const float*         emb      = (const float*)d_in[3];
  const float*         fc_bias  = (const float*)d_in[4];
  const float* Wq_t = (const float*)d_in[5];
  const float* Wk_t = (const float*)d_in[6];
  const float* Wv_t = (const float*)d_in[7];
  const float* Wo_t = (const float*)d_in[8];
  const float* g_t  = (const float*)d_in[9];
  const float* b_t  = (const float*)d_in[10];
  const float* Wq_s = (const float*)d_in[11];
  const float* Wk_s = (const float*)d_in[12];
  const float* Wv_s = (const float*)d_in[13];
  const float* Wo_s = (const float*)d_in[14];
  const float* g_s  = (const float*)d_in[15];
  const float* b_s  = (const float*)d_in[16];
  const float* W1   = (const float*)d_in[17];
  const float* b1   = (const float*)d_in[18];
  const float* W2   = (const float*)d_in[19];
  const float* b2   = (const float*)d_in[20];
  const float* g_f  = (const float*)d_in[21];
  const float* b_f  = (const float*)d_in[22];

  // Workspace layout (floats). Requires ~104 MB.
  const size_t MD = (size_t)MROWS * DMODEL;                 // 2,097,152
  float* ws    = (float*)d_ws;
  float* x     = ws;                                        // activations
  float* t1    = ws + 1 * MD;                               // q / tmp
  float* t2    = ws + 2 * MD;                               // k
  float* t3    = ws + 3 * MD;                               // v
  float* t4    = ws + 4 * MD;                               // attn out
  float* probs = ws + 5 * MD;                               // [B*H,LT,LS] 64 MB
  float* ffnh  = probs;                                     // aliased: [M, F] 32 MB

  float* logp = (float*)d_out;                              // [MROWS, VOCAB]
  float* cov  = logp + (size_t)MROWS * VOCAB;               // [B, LT, LS]

  const long sLTD = (long)LTGT * DMODEL;       // per-batch stride in [B,L,D]
  const long sPb  = (long)NHEAD * LTGT * LSRC; // per-batch stride in probs
  const long sPh  = (long)LTGT * LSRC;         // per-head stride in probs

  // 1) Embedding + positional encoding
  embed_pe_kernel<<<(MROWS * DMODEL) / 256, 256, 0, stream>>>(tgt, emb, x);

  for (int i = 0; i < NLAYER; ++i) {
    const float* wq = Wq_t + (size_t)i * DMODEL * DMODEL;
    const float* wk = Wk_t + (size_t)i * DMODEL * DMODEL;
    const float* wv = Wv_t + (size_t)i * DMODEL * DMODEL;
    const float* wo = Wo_t + (size_t)i * DMODEL * DMODEL;
    const float* wqs = Wq_s + (size_t)i * DMODEL * DMODEL;
    const float* wks = Wk_s + (size_t)i * DMODEL * DMODEL;
    const float* wvs = Wv_s + (size_t)i * DMODEL * DMODEL;
    const float* wos = Wo_s + (size_t)i * DMODEL * DMODEL;
    const float* w1 = W1 + (size_t)i * DMODEL * DFF;
    const float* w2 = W2 + (size_t)i * DFF * DMODEL;

    // ---- Self attention ----
    rungemm(stream, x, wq, nullptr, t1, MROWS, DMODEL, DMODEL, DMODEL, DMODEL, DMODEL, 1.f, false, false, 1, 1, 0,0,0,0,0,0);
    rungemm(stream, x, wk, nullptr, t2, MROWS, DMODEL, DMODEL, DMODEL, DMODEL, DMODEL, 1.f, false, false, 1, 1, 0,0,0,0,0,0);
    rungemm(stream, x, wv, nullptr, t3, MROWS, DMODEL, DMODEL, DMODEL, DMODEL, DMODEL, 1.f, false, false, 1, 1, 0,0,0,0,0,0);
    // S = Q Kt / sqrt(DH), per (b,h)
    rungemm(stream, t1, t2, nullptr, probs, LTGT, LTGT, DHEAD, DMODEL, DMODEL, LSRC,
            0.125f, true, false, BATCH * NHEAD, NHEAD, sLTD, DHEAD, sLTD, DHEAD, sPb, sPh);
    softmax_self_kernel<<<(BATCH * NHEAD * LTGT) / 8, 256, 0, stream>>>(probs, tgt);
    // O = P V, per (b,h); heads merge via ldc = D
    rungemm(stream, probs, t3, nullptr, t4, LTGT, DHEAD, LSRC, LSRC, DMODEL, DMODEL,
            1.f, false, false, BATCH * NHEAD, NHEAD, sPb, sPh, sLTD, DHEAD, sLTD, DHEAD);
    rungemm(stream, t4, wo, nullptr, t1, MROWS, DMODEL, DMODEL, DMODEL, DMODEL, DMODEL, 1.f, false, false, 1, 1, 0,0,0,0,0,0);
    lnres_kernel<<<MROWS / 8, 256, 0, stream>>>(t1, x, g_t + (size_t)i * DMODEL, b_t + (size_t)i * DMODEL, x);

    // ---- Cross attention ----
    rungemm(stream, x,       wqs, nullptr, t1, MROWS, DMODEL, DMODEL, DMODEL, DMODEL, DMODEL, 1.f, false, false, 1, 1, 0,0,0,0,0,0);
    rungemm(stream, context, wks, nullptr, t2, MROWS, DMODEL, DMODEL, DMODEL, DMODEL, DMODEL, 1.f, false, false, 1, 1, 0,0,0,0,0,0);
    rungemm(stream, context, wvs, nullptr, t3, MROWS, DMODEL, DMODEL, DMODEL, DMODEL, DMODEL, 1.f, false, false, 1, 1, 0,0,0,0,0,0);
    rungemm(stream, t1, t2, nullptr, probs, LTGT, LSRC, DHEAD, DMODEL, DMODEL, LSRC,
            0.125f, true, false, BATCH * NHEAD, NHEAD, sLTD, DHEAD, sLTD, DHEAD, sPb, sPh);
    softmax_cross_kernel<<<(BATCH * NHEAD * LTGT) / 8, 256, 0, stream>>>(probs, mask_src);
    if (i == NLAYER - 1)
      cov_kernel<<<(BATCH * LTGT * LSRC) / 256, 256, 0, stream>>>(probs, cov);
    rungemm(stream, probs, t3, nullptr, t4, LTGT, DHEAD, LSRC, LSRC, DMODEL, DMODEL,
            1.f, false, false, BATCH * NHEAD, NHEAD, sPb, sPh, sLTD, DHEAD, sLTD, DHEAD);
    rungemm(stream, t4, wos, nullptr, t1, MROWS, DMODEL, DMODEL, DMODEL, DMODEL, DMODEL, 1.f, false, false, 1, 1, 0,0,0,0,0,0);
    lnres_kernel<<<MROWS / 8, 256, 0, stream>>>(t1, x, g_s + (size_t)i * DMODEL, b_s + (size_t)i * DMODEL, x);

    // ---- FFN ----  (probs region is free now -> reuse as hidden)
    rungemm(stream, x, w1, b1 + (size_t)i * DFF, ffnh, MROWS, DFF, DMODEL, DMODEL, DFF, DFF,
            1.f, false, true, 1, 1, 0,0,0,0,0,0);
    rungemm(stream, ffnh, w2, b2 + (size_t)i * DMODEL, t1, MROWS, DMODEL, DFF, DFF, DMODEL, DMODEL,
            1.f, false, false, 1, 1, 0,0,0,0,0,0);
    lnres_kernel<<<MROWS / 8, 256, 0, stream>>>(t1, x, g_f + (size_t)i * DMODEL, b_f + (size_t)i * DMODEL, x);
  }

  // Logits = x @ emb^T + fc_bias, straight into d_out, then in-place
  // log-softmax (avoids a 524 MB scratch buffer).
  rungemm(stream, x, emb, fc_bias, logp, MROWS, VOCAB, DMODEL, DMODEL, DMODEL, VOCAB,
          1.f, true, false, 1, 1, 0,0,0,0,0,0);
  logsoftmax_kernel<<<MROWS, 256, 0, stream>>>(logp);
}
